// MultiHeadLatentAttention_66254165508249
// MI455X (gfx1250) — compile-verified, matrix operand order checked by reading, not run on hardware
//
#include <hip/hip_runtime.h>
#include <hip/hip_bf16.h>
#include <math.h>

// ---------------------------------------------------------------------------
// MLA attention, restructured for MI455X (gfx1250, wave32, WMMA):
//   C[.,0:12] = x@Wq_d + bq ; Ckv = x@Wkv_d + bkv             (WMMA f16, K=4096)
//   qs        = 0.5*(C_q@W_qk + b_qk)                          (VALU, K=12)
//   A[b,q,h,:4] = softmax(qs_h · Ckv^T) @ Ckv                  (flash, WMMA 16x16x4 f32)
//   out       = A @ W_eff + bias_eff,  W_eff = fold(Wv_u, Wo)  (WMMA f16, K=128)
// Rank-4 value folding turns ~206 GFLOP naive into ~10 GFLOP; total HBM
// traffic ~215 MB -> bandwidth bound (~9 us floor at 23.3 TB/s).
// B operands of both WMMA GEMMs are pre-packed into the exact per-lane VGPR
// layout (f16, 32B/lane/chunk) so the hot loops issue only b128 loads + wmma.
// The x-projection splits K into 4 segments (1024 waves in flight) so the
// 67 MB x stream can saturate HBM; partials reduced deterministically.
// WMMA loops use two independent accumulator chains to avoid the
// WMMA->VALU WAR hazard NOPs (ISA 7.12.1) and keep the XDL pipe busy.
// ---------------------------------------------------------------------------

#define D_MODEL 4096
#define NHEADS  32
#define HEAD_DIM 128
#define Q_LAT   12
#define KV_LAT  4
#define BATCH   2
#define SEQ     2048
#define BS      (BATCH * SEQ)      // 4096 rows total
#define NSEG    4                  // K-split for the projection GEMM
#define KSEG    (D_MODEL / NSEG)   // 1024
#define BSEG    32                 // row-segments for bias_eff reduction

typedef __attribute__((ext_vector_type(16))) _Float16 v16h;
typedef __attribute__((ext_vector_type(8)))  float    v8f;
typedef __attribute__((ext_vector_type(2)))  float    v2f;

#if defined(__has_builtin)
#if __has_builtin(__builtin_amdgcn_wmma_f32_16x16x4_f32)
#define HAS_WMMA_F32X4 1
#endif
#endif

// ---------------- kernel 0a: fold Wv_u into Wo ------------------------------
// W_eff[(h*4+d), n] = sum_dh Wv_u[d, h*128+dh] * Wo[h*128+dh, n]
__global__ __launch_bounds__(256) void k_weff(const float* __restrict__ Wv_u,
                                              const float* __restrict__ Wo,
                                              float* __restrict__ Weff) {
    int idx = blockIdx.x * blockDim.x + threadIdx.x;
    if (idx >= 128 * D_MODEL) return;
    int j = idx / D_MODEL;         // 0..127  (h*4 + d)
    int n = idx % D_MODEL;
    int h = j >> 2, d = j & 3;
    const float* wv = Wv_u + (size_t)d * (NHEADS * HEAD_DIM) + h * HEAD_DIM;
    const float* wo = Wo + (size_t)(h * HEAD_DIM) * D_MODEL + n;
    float s = 0.f;
    #pragma unroll 4
    for (int dh = 0; dh < HEAD_DIM; ++dh) s += wv[dh] * wo[(size_t)dh * D_MODEL];
    Weff[idx] = s;
}

// ---------------- kernel 0b: bias_eff = bv_u@Wo + bo (2-stage, coalesced) ---
__global__ __launch_bounds__(256) void k_beff_part(const float* __restrict__ bv_u,
                                                   const float* __restrict__ Wo,
                                                   float* __restrict__ bpart) {
    int t = blockIdx.x * blockDim.x + threadIdx.x;     // BSEG * D_MODEL
    if (t >= BSEG * D_MODEL) return;
    int seg = t >> 12;             // /4096
    int n   = t & (D_MODEL - 1);
    int r0  = seg * (NHEADS * HEAD_DIM / BSEG);        // 128 rows per segment
    float s = 0.f;
    #pragma unroll 4
    for (int r = 0; r < NHEADS * HEAD_DIM / BSEG; ++r)
        s += bv_u[r0 + r] * Wo[(size_t)(r0 + r) * D_MODEL + n];
    bpart[t] = s;
}

__global__ __launch_bounds__(256) void k_beff_fin(const float* __restrict__ bpart,
                                                  const float* __restrict__ bo,
                                                  float* __restrict__ beff) {
    int n = blockIdx.x * blockDim.x + threadIdx.x;
    if (n >= D_MODEL) return;
    float s = bo[n];
    #pragma unroll
    for (int seg = 0; seg < BSEG; ++seg) s += bpart[seg * D_MODEL + n];
    beff[n] = s;
}

// ---------------- kernel 0c: pack [Wq|Wkv] B-tiles into WMMA lane layout ----
// Record = (chunk c, lane): 16 f16 (32B).  Lane l: n=l&15, kb=8*(l>>4);
// elem i -> B(c*32+kb+i, n) ; elem i+8 -> B(c*32+16+kb+i, n)
__global__ __launch_bounds__(256) void k_packB(const float* __restrict__ Wq,
                                               const float* __restrict__ Wkv,
                                               _Float16* __restrict__ pB) {
    int t = blockIdx.x * blockDim.x + threadIdx.x;     // (D_MODEL/32)*32 = 4096
    if (t >= (D_MODEL / 32) * 32) return;
    int c = t >> 5, lane = t & 31;
    int lh = lane >> 4, ml = lane & 15, kb = lh * 8, n = ml;
    int k0 = c * 32;
    _Float16* dst = pB + (size_t)t * 16;
    #pragma unroll
    for (int i = 0; i < 8; ++i) {
        int k1 = k0 + kb + i;
        int k2 = k0 + 16 + kb + i;
        float b1 = (n < Q_LAT) ? Wq[(size_t)k1 * Q_LAT + n]
                               : Wkv[(size_t)k1 * KV_LAT + (n - Q_LAT)];
        float b2 = (n < Q_LAT) ? Wq[(size_t)k2 * Q_LAT + n]
                               : Wkv[(size_t)k2 * KV_LAT + (n - Q_LAT)];
        dst[i]     = (_Float16)b1;
        dst[i + 8] = (_Float16)b2;
    }
}

// ---------------- kernel 0d: pack W_eff B-tiles into WMMA lane layout -------
// Record = (tn, chunk c, lane): 16 f16 (32B).  n = tn*16 + (l&15).
__global__ __launch_bounds__(256) void k_packW(const float* __restrict__ Weff,
                                               _Float16* __restrict__ pW) {
    int t = blockIdx.x * blockDim.x + threadIdx.x;     // 256*4*32 = 32768
    if (t >= (D_MODEL / 16) * 4 * 32) return;
    int tn = t >> 7;                // /(4*32)
    int c  = (t >> 5) & 3;
    int lane = t & 31;
    int lh = lane >> 4, ml = lane & 15, kb = lh * 8;
    int n = tn * 16 + ml;
    int k0 = c * 32;
    _Float16* dst = pW + (size_t)t * 16;
    #pragma unroll
    for (int i = 0; i < 8; ++i) {
        dst[i]     = (_Float16)Weff[(size_t)(k0 + kb + i) * D_MODEL + n];
        dst[i + 8] = (_Float16)Weff[(size_t)(k0 + 16 + kb + i) * D_MODEL + n];
    }
}

// ---------------- kernel 1: latent projections via WMMA (K-split x4) --------
// Wave (seg, tile): 16-row tile, K range [seg*1024, seg*1024+1024).
// N=16 (12 Cq cols + 4 Ckv cols) in one WMMA tile; f16 chunks of 32.
// Two accumulator chains (even/odd chunk) avoid WMMA hazard NOPs.
__global__ __launch_bounds__(256) void k_proj(const float* __restrict__ x,
                                              const _Float16* __restrict__ pB,
                                              float* __restrict__ Cpart) {
    int wid  = (blockIdx.x * blockDim.x + threadIdx.x) >> 5;   // 0..1023
    int lane = threadIdx.x & 31;
    int seg  = wid >> 8;            // 0..3
    int tile = wid & 255;
    int row0 = tile * 16;
    int lh = lane >> 4;             // half-wave (0/1)
    int ml = lane & 15;             // M for A-tile / N for B-tile
    int kb = lh * 8;                // per-half K base within 32-wide chunk
    const float* xrow = x + (size_t)(row0 + ml) * D_MODEL;
    int kbeg = seg * KSEG;
    v8f acc0 = {}, acc1 = {};
    for (int k0 = kbeg; k0 < kbeg + KSEG; k0 += 64) {
        const float* xr = xrow + k0;
        __builtin_prefetch(xr + 128, 0, 1);            // global_prefetch_b8
        v16h a0, a1;
        #pragma unroll
        for (int i = 0; i < 8; ++i) {
            a0[i]     = (_Float16)xr[kb + i];          // K = kb..kb+7
            a0[i + 8] = (_Float16)xr[16 + kb + i];     // K = 16+kb..16+kb+7
            a1[i]     = (_Float16)xr[32 + kb + i];
            a1[i + 8] = (_Float16)xr[48 + kb + i];
        }
        const _Float16* pr = pB + ((size_t)(k0 >> 5) * 32 + lane) * 16;
        v16h b0 = *(const v16h*)pr;
        v16h b1 = *(const v16h*)(pr + 32 * 16);
        acc0 = __builtin_amdgcn_wmma_f32_16x16x32_f16(false, a0, false, b0,
                                                      (short)0, acc0, false, false);
        acc1 = __builtin_amdgcn_wmma_f32_16x16x32_f16(false, a1, false, b1,
                                                      (short)0, acc1, false, false);
    }
    float* dst = Cpart + (size_t)seg * BS * 16;
    #pragma unroll
    for (int v = 0; v < 8; ++v)
        dst[(size_t)(row0 + v + 8 * lh) * 16 + ml] = acc0[v] + acc1[v];
}

// ---------------- kernel 1b: reduce K-partials, add bias, emit C & Ckv ------
__global__ __launch_bounds__(256) void k_reduceC(const float* __restrict__ Cpart,
                                                 const float* __restrict__ bq,
                                                 const float* __restrict__ bkv,
                                                 float* __restrict__ C,
                                                 float* __restrict__ Ckv) {
    int idx = blockIdx.x * blockDim.x + threadIdx.x;   // BS*16
    if (idx >= BS * 16) return;
    int n = idx & 15, row = idx >> 4;
    float s = (n < Q_LAT) ? bq[n] : bkv[n - Q_LAT];
    #pragma unroll
    for (int seg = 0; seg < NSEG; ++seg) s += Cpart[(size_t)seg * BS * 16 + idx];
    C[idx] = s;
    if (n >= Q_LAT) Ckv[(size_t)row * 4 + (n - Q_LAT)] = s;
}

// ---------------- kernel 2: q latent -> per-head kv-latent query ------------
// qs[row, h*4+d] = 0.5 * (C_q[row,:] @ W_qk[:, h*4+d] + b_qk). K=12 -> VALU.
__global__ __launch_bounds__(256) void k_q(const float* __restrict__ C,
                                           const float* __restrict__ Wqk,
                                           const float* __restrict__ bqk,
                                           float* __restrict__ qs) {
    int idx = blockIdx.x * blockDim.x + threadIdx.x;
    if (idx >= BS * 128) return;
    int row = idx >> 7;
    int n   = idx & 127;
    const float* cq = C + (size_t)row * 16;
    float s = bqk[n];
    #pragma unroll
    for (int j = 0; j < Q_LAT; ++j) s += cq[j] * Wqk[j * 128 + n];
    qs[idx] = 0.5f * s;            // fold 1/sqrt(KV_LAT) into q
}

// ---------------- kernel 3: fused flash attention (rank-4 K and V) ----------
// One wave per (b, h, 16-query tile). Scores via V_WMMA_F32_16X16X4_F32;
// online softmax over 16-key tiles; accumulates attn@Ckv ([16 x 4]).
__global__ __launch_bounds__(32) void k_attn(const float* __restrict__ Ckv,
                                             const float* __restrict__ qs,
                                             float* __restrict__ Aout) {
    int bid  = blockIdx.x;                 // B * H * (SEQ/16)
    int lane = threadIdx.x & 31;
    int qt = bid & 127;
    int h  = (bid >> 7) & 31;
    int b  = bid >> 12;
    int lh = lane >> 4, ml = lane & 15;
    int q0 = b * SEQ + qt * 16;            // global query row of this tile

#ifdef HAS_WMMA_F32X4
    // A-tile (16x4 f32): lane = M (m=ml), halves carry K pairs {0,1}/{2,3}
    v2f aq;
    {
        const float* qr = qs + (size_t)(q0 + ml) * 128 + h * 4 + 2 * lh;
        aq.x = qr[0];
        aq.y = qr[1];
    }
#else
    float qrow[8][4];                      // rows v+8*lh of the q tile
    #pragma unroll
    for (int v = 0; v < 8; ++v) {
        const float* qr = qs + (size_t)(q0 + v + 8 * lh) * 128 + h * 4;
        qrow[v][0] = qr[0]; qrow[v][1] = qr[1];
        qrow[v][2] = qr[2]; qrow[v][3] = qr[3];
    }
#endif

    float Mr[8], Lr[8], acc[8][4];
    #pragma unroll
    for (int v = 0; v < 8; ++v) {
        Mr[v] = -INFINITY; Lr[v] = 0.f;
        acc[v][0] = acc[v][1] = acc[v][2] = acc[v][3] = 0.f;
    }

    for (int kt = 0; kt < SEQ; kt += 16) {
        int krow = b * SEQ + kt;
        const float* ck = Ckv + (size_t)(krow + ml) * 4;       // this lane's key
        float4 ckv = *(const float4*)ck;                       // 16B aligned

#ifdef HAS_WMMA_F32X4
        // B-tile (4x16 f32): lane = N (key ml), halves carry K pairs
        v2f bkv; bkv.x = ck[2 * lh]; bkv.y = ck[2 * lh + 1];
        v8f zero = {};
        v8f sc = __builtin_amdgcn_wmma_f32_16x16x4_f32(false, aq, false, bkv,
                                                       (short)0, zero, false, false);
#else
        v8f sc;
        #pragma unroll
        for (int v = 0; v < 8; ++v)
            sc[v] = qrow[v][0] * ckv.x + qrow[v][1] * ckv.y +
                    qrow[v][2] * ckv.z + qrow[v][3] * ckv.w;
#endif
        // Online softmax per row; D layout: lane=col(key), VGPR v = row v+8*lh.
        #pragma unroll
        for (int v = 0; v < 8; ++v) {
            float s = sc[v];
            float m = s;
            m = fmaxf(m, __shfl_xor(m, 1, 32));
            m = fmaxf(m, __shfl_xor(m, 2, 32));
            m = fmaxf(m, __shfl_xor(m, 4, 32));
            m = fmaxf(m, __shfl_xor(m, 8, 32));   // row max within 16-lane half
            float mnew  = fmaxf(Mr[v], m);
            float scale = __expf(Mr[v] - mnew);   // 0 on first tile (exp(-inf))
            float p     = __expf(s - mnew);
            Mr[v] = mnew;
            Lr[v] = Lr[v] * scale + p;
            acc[v][0] = acc[v][0] * scale + p * ckv.x;
            acc[v][1] = acc[v][1] * scale + p * ckv.y;
            acc[v][2] = acc[v][2] * scale + p * ckv.z;
            acc[v][3] = acc[v][3] * scale + p * ckv.w;
        }
    }

    // Reduce per-lane partial sums across the 16 keys/lanes of each half.
    #pragma unroll
    for (int v = 0; v < 8; ++v) {
        float L  = Lr[v];
        float a0 = acc[v][0], a1 = acc[v][1], a2 = acc[v][2], a3 = acc[v][3];
        #pragma unroll
        for (int off = 1; off <= 8; off <<= 1) {
            L  += __shfl_xor(L,  off, 32);
            a0 += __shfl_xor(a0, off, 32);
            a1 += __shfl_xor(a1, off, 32);
            a2 += __shfl_xor(a2, off, 32);
            a3 += __shfl_xor(a3, off, 32);
        }
        if (ml == 0) {
            float inv = 1.0f / L;
            int row = q0 + v + 8 * lh;
            float4 o = make_float4(a0 * inv, a1 * inv, a2 * inv, a3 * inv);
            *(float4*)(Aout + (size_t)row * 128 + h * 4) = o;
        }
    }
}

// ---------------- kernel 4: out = A @ W_eff + bias_eff (WMMA f16, K=128) ----
// B-tiles come pre-packed; one wave per 16x16 output tile, 4 WMMA in two
// independent accumulator chains.
__global__ __launch_bounds__(256) void k_out(const float* __restrict__ Aout,
                                             const _Float16* __restrict__ pW,
                                             const float* __restrict__ beff,
                                             float* __restrict__ out) {
    int wid  = (blockIdx.x * blockDim.x + threadIdx.x) >> 5;
    int lane = threadIdx.x & 31;
    const int tiles_n = D_MODEL / 16;      // 256
    int tm = wid / tiles_n, tn = wid % tiles_n;
    if (tm >= BS / 16) return;
    int lh = lane >> 4, ml = lane & 15;
    int kb = lh * 8;
    v8f acc0 = {}, acc1 = {};
    #pragma unroll
    for (int c = 0; c < 4; c += 2) {
        v16h a0, a1;
        const float* ar = Aout + (size_t)(tm * 16 + ml) * 128 + c * 32;
        #pragma unroll
        for (int i = 0; i < 8; ++i) {
            a0[i]     = (_Float16)ar[kb + i];
            a0[i + 8] = (_Float16)ar[16 + kb + i];
            a1[i]     = (_Float16)ar[32 + kb + i];
            a1[i + 8] = (_Float16)ar[48 + kb + i];
        }
        const _Float16* pr = pW + (((size_t)tn * 4 + c) * 32 + lane) * 16;
        v16h b0 = *(const v16h*)pr;
        v16h b1 = *(const v16h*)(pr + 32 * 16);
        acc0 = __builtin_amdgcn_wmma_f32_16x16x32_f16(false, a0, false, b0,
                                                      (short)0, acc0, false, false);
        acc1 = __builtin_amdgcn_wmma_f32_16x16x32_f16(false, a1, false, b1,
                                                      (short)0, acc1, false, false);
    }
    int n = tn * 16 + ml;
    float bias = beff[n];
    #pragma unroll
    for (int v = 0; v < 8; ++v)
        out[(size_t)(tm * 16 + v + 8 * lh) * D_MODEL + n] = acc0[v] + acc1[v] + bias;
}

// ---------------------------------------------------------------------------
extern "C" void kernel_launch(void* const* d_in, const int* in_sizes, int n_in,
                              void* d_out, int out_size, void* d_ws, size_t ws_size,
                              hipStream_t stream) {
    const float* x    = (const float*)d_in[0];
    const float* Wq   = (const float*)d_in[1];
    const float* bq   = (const float*)d_in[2];
    const float* Wqk  = (const float*)d_in[3];
    const float* bqk  = (const float*)d_in[4];
    const float* Wkv  = (const float*)d_in[5];
    const float* bkv  = (const float*)d_in[6];
    const float* Wv_u = (const float*)d_in[7];
    const float* bv_u = (const float*)d_in[8];
    const float* Wo   = (const float*)d_in[9];
    const float* bo   = (const float*)d_in[10];
    float* out = (float*)d_out;

    // workspace layout (all offsets 16B-aligned)
    float* ws    = (float*)d_ws;
    float* C     = ws;                            // [4096, 16]
    float* qs    = C     + (size_t)BS * 16;       // [4096, 128]
    float* Aout  = qs    + (size_t)BS * 128;      // [4096, 128]
    float* Weff  = Aout  + (size_t)BS * 128;      // [128, 4096]
    float* beff  = Weff  + (size_t)128 * D_MODEL; // [4096]
    float* Ckv   = beff  + D_MODEL;               // [4096, 4]
    float* Cpart = Ckv   + (size_t)BS * 4;        // [NSEG][4096, 16]
    float* bpart = Cpart + (size_t)NSEG * BS * 16;// [BSEG][4096]
    _Float16* pB = (_Float16*)(bpart + (size_t)BSEG * D_MODEL);  // [128][32][16]
    _Float16* pW = pB + (size_t)(D_MODEL / 32) * 32 * 16;        // [256][4][32][16]

    k_weff<<<(128 * D_MODEL + 255) / 256, 256, 0, stream>>>(Wv_u, Wo, Weff);
    k_beff_part<<<(BSEG * D_MODEL + 255) / 256, 256, 0, stream>>>(bv_u, Wo, bpart);
    k_beff_fin<<<(D_MODEL + 255) / 256, 256, 0, stream>>>(bpart, bo, beff);
    k_packB<<<((D_MODEL / 32) * 32 + 255) / 256, 256, 0, stream>>>(Wq, Wkv, pB);
    k_packW<<<((D_MODEL / 16) * 4 * 32 + 255) / 256, 256, 0, stream>>>(Weff, pW);
    k_proj<<<NSEG * (BS / 16) / 8, 256, 0, stream>>>(x, pB, Cpart);
    k_reduceC<<<(BS * 16 + 255) / 256, 256, 0, stream>>>(Cpart, bq, bkv, C, Ckv);
    k_q<<<(BS * 128 + 255) / 256, 256, 0, stream>>>(C, Wqk, bqk, qs);
    k_attn<<<BATCH * NHEADS * (SEQ / 16), 32, 0, stream>>>(Ckv, qs, Aout);
    k_out<<<(BS / 16) * (D_MODEL / 16) / 8, 256, 0, stream>>>(Aout, pW, beff, out);
}
